// DINOv3ViTAttention_74045236183510
// MI455X (gfx1250) — matrix operand order repaired and checked by reading, not compiled, not run on hardware
//
#include <hip/hip_runtime.h>
#include <hip/hip_bf16.h>

typedef __attribute__((ext_vector_type(16))) __bf16 bf16x16;
typedef __attribute__((ext_vector_type(8)))  __bf16 bf16x8;
typedef __attribute__((ext_vector_type(8)))  float  f32x8;
typedef unsigned short u16;

#define B_   8
#define S_   1129
#define D_   768
#define H_   12
#define SPAD 1152
#define M_   (B_ * S_)   // 9032

// ---------- helpers ----------

__device__ __forceinline__ u16 f2bf(float f) {
  unsigned u = __builtin_bit_cast(unsigned, f);
  u += 0x7FFFu + ((u >> 16) & 1u);          // round-to-nearest-even
  return (u16)(u >> 16);
}

__device__ __forceinline__ bf16x16 cat8(bf16x8 lo, bf16x8 hi) {
  return __builtin_shufflevector(lo, hi, 0,1,2,3,4,5,6,7,8,9,10,11,12,13,14,15);
}

// 16 contiguous bf16 (B-fragment)
__device__ __forceinline__ bf16x16 load16_bf(const u16* p) {
  bf16x8 lo = *(const bf16x8*)p;
  bf16x8 hi = *(const bf16x8*)(p + 8);
  return cat8(lo, hi);
}

// A-fragment: 16 contiguous elements at p and p+8 (contiguous k order)
__device__ __forceinline__ bf16x16 load_a_bf(const u16* p) {
  bf16x8 lo = *(const bf16x8*)p;
  bf16x8 hi = *(const bf16x8*)(p + 8);
  return cat8(lo, hi);
}

__device__ __forceinline__ f32x8 wmma_bf16(bf16x16 a, bf16x16 b, f32x8 c) {
  return __builtin_amdgcn_wmma_f32_16x16x32_bf16(false, a, false, b, (short)0, c,
                                                 false, false);
}

// ---------- kernel 0: f32 -> bf16 pre-conversion ----------

__global__ __launch_bounds__(256)
void cvt_bf16_kernel(const float* __restrict__ src, u16* __restrict__ dst, int n) {
  int i = (blockIdx.x * 256 + threadIdx.x) * 4;
  const int stride = gridDim.x * 256 * 4;
  for (; i + 3 < n; i += stride) {
    dst[i + 0] = f2bf(src[i + 0]);
    dst[i + 1] = f2bf(src[i + 1]);
    dst[i + 2] = f2bf(src[i + 2]);
    dst[i + 3] = f2bf(src[i + 3]);
  }
  for (; i < n; ++i) dst[i] = f2bf(src[i]);
}

// ---------- kernel 1: QKV projection + RoPE / det-scale + layout ----------
// grid = (ceil(M/64), 12 heads, 3 {q,k,v}), block = 256 (8 waves: 4M x 2N)

__global__ __launch_bounds__(256)
void qkv_proj_kernel(const u16* __restrict__ hsbf,
                     const u16* __restrict__ wqb, const u16* __restrict__ wkb,
                     const u16* __restrict__ wvb,
                     const float* __restrict__ qb, const float* __restrict__ vb,
                     const float* __restrict__ cosp, const float* __restrict__ sinp,
                     const float* __restrict__ ph,  const float* __restrict__ pwd,
                     const int* __restrict__ rs_p, const int* __restrict__ re_p,
                     const int* __restrict__ ds_p, const int* __restrict__ de_p,
                     u16* __restrict__ qbf, u16* __restrict__ kbf,
                     u16* __restrict__ vT) {
  __shared__ float tile[64][65];

  const int z = blockIdx.z;
  const u16*   W    = (z == 0) ? wqb : (z == 1) ? wkb : wvb;
  const float* bias = (z == 0) ? qb : (z == 2) ? vb : nullptr;

  const int tid  = threadIdx.x;
  const int w    = tid >> 5, lane = tid & 31;
  const int hf   = lane >> 4, lcol = lane & 15;
  const int wm   = w >> 1,   wn   = w & 1;
  const int mbase = blockIdx.x * 64;
  const int nbase = blockIdx.y * 64;   // head * 64

  int arow = mbase + wm * 16 + lcol;
  if (arow >= M_) arow = M_ - 1;                 // clamp tail rows
  const u16* aptr = hsbf + (size_t)arow * D_ + (hf ? 16 : 0);

  const int n0 = nbase + wn * 32 + lcol;
  const u16* w0 = W + (size_t)n0 * D_ + (hf ? 16 : 0);
  const u16* w1 = W + (size_t)(n0 + 16) * D_ + (hf ? 16 : 0);

  f32x8 acc0 = {0.f,0.f,0.f,0.f,0.f,0.f,0.f,0.f};
  f32x8 acc1 = {0.f,0.f,0.f,0.f,0.f,0.f,0.f,0.f};

  // fully unrolled k-loop: fresh registers per step, scheduler hoists loads
#pragma unroll
  for (int kb = 0; kb < D_; kb += 32) {
    bf16x16 a = load_a_bf(aptr + kb);
    acc0 = wmma_bf16(a, load16_bf(w0 + kb), acc0);
    acc1 = wmma_bf16(a, load16_bf(w1 + kb), acc1);
  }

  const float b0v = bias ? bias[n0]      : 0.f;
  const float b1v = bias ? bias[n0 + 16] : 0.f;
#pragma unroll
  for (int r = 0; r < 8; ++r) {
    const int rl = wm * 16 + r + hf * 8;
    tile[rl][wn * 32 + lcol]      = acc0[r] + b0v;
    tile[rl][wn * 32 + 16 + lcol] = acc1[r] + b1v;
  }
  __syncthreads();

  // epilogue: 256 threads, 16 elements each, over the 64x64 f32 tile
  const int rs = rs_p[0], re = re_p[0], dst = ds_p[0], de = de_p[0];
  const int row = tid >> 2;
  const int cb  = (tid & 3) * 16;
  const int m   = mbase + row;
  if (m < M_) {
    const int bb = m / S_, s = m % S_;
    const int h  = blockIdx.y;
#pragma unroll
    for (int i = 0; i < 16; ++i) {
      const int col = cb + i;
      const float x = tile[row][col];
      if (z == 2) {
        vT[(((size_t)bb * H_ + h) * 64 + col) * SPAD + s] = f2bf(x);
      } else {
        float xo = x;
        if (s >= rs && s < re) {                       // RoPE
          const int pos = s - rs;
          const float c  = cosp[pos * 64 + col];
          const float sn = sinp[pos * 64 + col];
          const float p  = tile[row][col ^ 32];
          xo = x * c + (col < 32 ? -p : p) * sn;
        } else if (s >= dst && s < de) {               // det scale
          const int i2 = s - dst;
          xo = x * __expf(2.0f * (col < 32 ? ph[i2] : pwd[i2]));
        }
        u16* d = (z == 0) ? qbf : kbf;
        d[(((size_t)bb * H_ + h) * SPAD + s) * 64 + col] = f2bf(xo);
      }
    }
  }
}

// ---------- kernel 2: flash attention ----------
// grid = (SPAD/128, 12, 8), block = 256 (8 waves, 16 query rows each)

__global__ __launch_bounds__(256)
void attn_kernel(const u16* __restrict__ qbf, const u16* __restrict__ kbf,
                 const u16* __restrict__ vT,  const float* __restrict__ gate,
                 u16* __restrict__ ctx) {
  __shared__ __attribute__((aligned(16))) u16 plds[8 * 16 * 32];

  const int tid  = threadIdx.x;
  const int w    = tid >> 5, lane = tid & 31;
  const int hf   = lane >> 4, lcol = lane & 15;
  const int h    = blockIdx.y, bb = blockIdx.z;
  const int bh   = bb * H_ + h;
  const int qr0  = blockIdx.x * 128 + w * 16;

  // clamped gate row pointers (tail query rows are masked at the store)
  const float* grow[8];
#pragma unroll
  for (int r = 0; r < 8; ++r) {
    int qr = qr0 + r + hf * 8;
    if (qr >= S_) qr = S_ - 1;
    grow[r] = gate + (size_t)qr * S_;
  }

  // Q A-fragments (kept in registers for the whole key loop)
  const u16* qrowp = qbf + ((size_t)bh * SPAD + (qr0 + lcol)) * 64;
  const bf16x16 qf0 = load_a_bf(qrowp + 0  + (hf ? 16 : 0));
  const bf16x16 qf1 = load_a_bf(qrowp + 32 + (hf ? 16 : 0));

  const f32x8 zero = {0.f,0.f,0.f,0.f,0.f,0.f,0.f,0.f};
  f32x8 oa[4];
#pragma unroll
  for (int nt = 0; nt < 4; ++nt) oa[nt] = zero;

  float mrow[8], lrow[8];
#pragma unroll
  for (int r = 0; r < 8; ++r) { mrow[r] = -3.0e38f; lrow[r] = 0.f; }

  u16* pl = plds + w * 512;
  const u16* vbase = vT + (size_t)bh * 64 * SPAD;

  for (int jb = 0; jb < SPAD; jb += 32) {
    if (jb + 32 < SPAD)   // pull next K tile toward the caches
      __builtin_prefetch(kbf + ((size_t)bh * SPAD + jb + 32 + lane) * 64, 0, 1);

    // S = Q @ K^T, scale, gate, mask
    f32x8 sf[2];
#pragma unroll
    for (int ns = 0; ns < 2; ++ns) {
      f32x8 c = zero;
      const int key  = jb + ns * 16 + lcol;
      const int keyc = (key < S_) ? key : (S_ - 1);       // clamp for gate load
      const u16* kp  = kbf + ((size_t)bh * SPAD + key) * 64;
      c = wmma_bf16(qf0, load16_bf(kp + 0  + (hf ? 16 : 0)), c);
      c = wmma_bf16(qf1, load16_bf(kp + 32 + (hf ? 16 : 0)), c);
#pragma unroll
      for (int r = 0; r < 8; ++r) {
        float sv = c[r] * 0.125f + grow[r][keyc];         // hd^-0.5 = 1/8
        if (key >= S_) sv = -3.0e38f;
        c[r] = sv;
      }
      sf[ns] = c;
    }

    // online softmax: row max / alpha / P / row sum (16-lane butterflies)
    float alpha[8];
#pragma unroll
    for (int r = 0; r < 8; ++r) {
      float t = fmaxf(sf[0][r], sf[1][r]);
      t = fmaxf(t, __shfl_xor(t, 1, 32));
      t = fmaxf(t, __shfl_xor(t, 2, 32));
      t = fmaxf(t, __shfl_xor(t, 4, 32));
      t = fmaxf(t, __shfl_xor(t, 8, 32));
      const float nm = fmaxf(mrow[r], t);
      alpha[r] = __expf(mrow[r] - nm);
      mrow[r]  = nm;
    }
#pragma unroll
    for (int r = 0; r < 8; ++r) {
      const float p0 = __expf(sf[0][r] - mrow[r]);
      const float p1 = __expf(sf[1][r] - mrow[r]);
      sf[0][r] = p0; sf[1][r] = p1;
      float t = p0 + p1;
      t += __shfl_xor(t, 1, 32);
      t += __shfl_xor(t, 2, 32);
      t += __shfl_xor(t, 4, 32);
      t += __shfl_xor(t, 8, 32);
      lrow[r] = lrow[r] * alpha[r] + t;
    }

    // rescale O; stage P (C-layout -> A-layout) through wave-private LDS
#pragma unroll
    for (int r = 0; r < 8; ++r) {
#pragma unroll
      for (int nt = 0; nt < 4; ++nt) oa[nt][r] *= alpha[r];
      pl[(r + hf * 8) * 32 + lcol]      = f2bf(sf[0][r]);
      pl[(r + hf * 8) * 32 + 16 + lcol] = f2bf(sf[1][r]);
    }
    const bf16x16 pa = load_a_bf(pl + lcol * 32 + (hf ? 16 : 0));

    // O += P @ V   (V stored transposed -> contiguous B-fragments)
#pragma unroll
    for (int nt = 0; nt < 4; ++nt) {
      const u16* vp = vbase + (size_t)(nt * 16 + lcol) * SPAD + jb + (hf ? 16 : 0);
      oa[nt] = wmma_bf16(pa, load16_bf(vp), oa[nt]);
    }
  }

  // write ctx (bf16, [B, S, H*64] for the output GEMM)
#pragma unroll
  for (int r = 0; r < 8; ++r) {
    const int qr = qr0 + r + hf * 8;
    if (qr < S_) {
      const float inv = 1.0f / lrow[r];
      const size_t base = ((size_t)bb * S_ + qr) * D_ + h * 64;
#pragma unroll
      for (int nt = 0; nt < 4; ++nt)
        ctx[base + nt * 16 + lcol] = f2bf(oa[nt][r] * inv);
    }
  }
}

// ---------- kernel 3: output projection ----------
// grid = (ceil(M/64), 12), block = 256

__global__ __launch_bounds__(256)
void oproj_kernel(const u16* __restrict__ ctx, const u16* __restrict__ owb,
                  const float* __restrict__ ob, float* __restrict__ out) {
  const int tid  = threadIdx.x;
  const int w    = tid >> 5, lane = tid & 31;
  const int hf   = lane >> 4, lcol = lane & 15;
  const int wm   = w >> 1,   wn   = w & 1;
  const int mbase = blockIdx.x * 64, nbase = blockIdx.y * 64;

  int arow = mbase + wm * 16 + lcol;
  if (arow >= M_) arow = M_ - 1;
  const u16* ap = ctx + (size_t)arow * D_ + (hf ? 16 : 0);

  const int n0 = nbase + wn * 32 + lcol;
  const u16* w0 = owb + (size_t)n0 * D_ + (hf ? 16 : 0);
  const u16* w1 = owb + (size_t)(n0 + 16) * D_ + (hf ? 16 : 0);

  f32x8 acc0 = {0.f,0.f,0.f,0.f,0.f,0.f,0.f,0.f};
  f32x8 acc1 = {0.f,0.f,0.f,0.f,0.f,0.f,0.f,0.f};

#pragma unroll
  for (int kb = 0; kb < D_; kb += 32) {
    bf16x16 a = load_a_bf(ap + kb);
    acc0 = wmma_bf16(a, load16_bf(w0 + kb), acc0);
    acc1 = wmma_bf16(a, load16_bf(w1 + kb), acc1);
  }

  const float b0v = ob[n0], b1v = ob[n0 + 16];
#pragma unroll
  for (int r = 0; r < 8; ++r) {
    const int m = mbase + wm * 16 + r + hf * 8;
    if (m < M_) {
      out[(size_t)m * D_ + n0]      = acc0[r] + b0v;
      out[(size_t)m * D_ + n0 + 16] = acc1[r] + b1v;
    }
  }
}

// ---------- launch ----------

extern "C" void kernel_launch(void* const* d_in, const int* in_sizes, int n_in,
                              void* d_out, int out_size, void* d_ws, size_t ws_size,
                              hipStream_t stream) {
  (void)in_sizes; (void)n_in; (void)out_size; (void)ws_size;

  const float* hs   = (const float*)d_in[0];
  const float* qw   = (const float*)d_in[1];
  const float* qb   = (const float*)d_in[2];
  const float* kw   = (const float*)d_in[3];
  const float* vw   = (const float*)d_in[4];
  const float* vb   = (const float*)d_in[5];
  const float* ow   = (const float*)d_in[6];
  const float* ob   = (const float*)d_in[7];
  const float* cosp = (const float*)d_in[8];
  const float* sinp = (const float*)d_in[9];
  const float* ph   = (const float*)d_in[10];
  const float* pwd  = (const float*)d_in[11];
  const float* gate = (const float*)d_in[12];
  const int* rs = (const int*)d_in[13];
  const int* re = (const int*)d_in[14];
  const int* ds = (const int*)d_in[15];
  const int* de = (const int*)d_in[16];

  char* ws = (char*)d_ws;
  const size_t sz_head = (size_t)B_ * H_ * SPAD * 64 * sizeof(u16);  // 14.2 MB
  const size_t sz_ctx  = (size_t)M_ * D_ * sizeof(u16);              // 13.9 MB
  const size_t sz_hs   = (size_t)M_ * D_ * sizeof(u16);              // 13.9 MB
  const size_t sz_w    = (size_t)D_ * D_ * sizeof(u16);              // 1.2 MB

  u16* qbf  = (u16*)(ws);
  u16* kbf  = (u16*)(ws + sz_head);
  u16* vT   = (u16*)(ws + 2 * sz_head);
  u16* ctx  = (u16*)(ws + 3 * sz_head);
  u16* hsbf = (u16*)(ws + 3 * sz_head + sz_ctx);
  u16* wqb  = (u16*)(ws + 3 * sz_head + sz_ctx + sz_hs);
  u16* wkb  = (u16*)((char*)wqb + sz_w);
  u16* wvb  = (u16*)((char*)wkb + sz_w);
  u16* owb  = (u16*)((char*)wvb + sz_w);

  // pre-convert activations + weights to bf16 (one pass, L2 resident after)
  const int nhs = M_ * D_, nw = D_ * D_;
  cvt_bf16_kernel<<<dim3((nhs / 4 + 255) / 256), 256, 0, stream>>>(hs, hsbf, nhs);
  cvt_bf16_kernel<<<dim3((nw  / 4 + 255) / 256), 256, 0, stream>>>(qw, wqb, nw);
  cvt_bf16_kernel<<<dim3((nw  / 4 + 255) / 256), 256, 0, stream>>>(kw, wkb, nw);
  cvt_bf16_kernel<<<dim3((nw  / 4 + 255) / 256), 256, 0, stream>>>(vw, wvb, nw);
  cvt_bf16_kernel<<<dim3((nw  / 4 + 255) / 256), 256, 0, stream>>>(ow, owb, nw);

  dim3 g1((M_ + 63) / 64, H_, 3);
  qkv_proj_kernel<<<g1, 256, 0, stream>>>(hsbf, wqb, wkb, wvb, qb, vb, cosp, sinp,
                                          ph, pwd, rs, re, ds, de, qbf, kbf, vT);
  dim3 g2(SPAD / 128, H_, B_);
  attn_kernel<<<g2, 256, 0, stream>>>(qbf, kbf, vT, gate, ctx);

  dim3 g3((M_ + 63) / 64, D_ / 64, 1);
  oproj_kernel<<<g3, 256, 0, stream>>>(ctx, owb, ob, (float*)d_out);
}
